// FlowNet_55327768708603
// MI455X (gfx1250) — matrix operand order repaired and checked by reading, not compiled, not run on hardware
//
#include <hip/hip_runtime.h>

typedef __attribute__((ext_vector_type(16))) _Float16 v16h;
typedef __attribute__((ext_vector_type(8)))  _Float16 v8h;
typedef __attribute__((ext_vector_type(8)))  float    v8f;

#define LEAKY_NEG 0.1f
#define WN_OUT 16
#define GEMM_WAVES 8   // 8 waves/block: 8 M-tiles share one LDS-staged B tile

static __device__ __forceinline__ float leaky1(float x) {
  return x >= 0.0f ? x : LEAKY_NEG * x;
}

// ---------------------------------------------------------------------------
// GEMM: Y[M,N] = X[M,K] @ W[N,K]^T + bias[N]   (f32 in/out, f16 WMMA core)
// Block = 256 threads = 8 waves. Wave w computes tile (blockIdx.x*128 + w*16,
// blockIdx.y*16). The W tile (16 cols x 32 k, f32) is loaded coalesced by the
// whole block, converted to f16 into LDS once, and read back by every wave as
// two aligned ds_load_b128 per fragment. A operands are loaded per lane with
// clamped row pointers (no EXEC divergence in the main K loop); K tail is a
// single guarded zero-filled step.
// ---------------------------------------------------------------------------
__global__ void k_gemm_wmma(const float* __restrict__ X, const float* __restrict__ W,
                            const float* __restrict__ bias, float* __restrict__ Y,
                            int M, int N, int K) {
  __shared__ __align__(16) _Float16 lb[16 * 32];   // [n within tile][k within step]

  const int t    = threadIdx.x;
  const int wave = t >> 5;
  const int lane = t & 31;
  const int hs   = lane >> 4;      // half-wave select
  const int li   = lane & 15;
  const int tm   = blockIdx.x * (16 * GEMM_WAVES) + wave * 16;
  const int tn   = blockIdx.y << 4;

  const int arow = tm + li;
  const int bcol = tn + li;
  const bool aok = arow < M;
  const bool bok = bcol < N;
  // Clamped pointers: out-of-range rows/cols compute garbage that is never
  // stored (row r of D depends only on row r of A; col n only on col n of B).
  const float* xrow = X + (size_t)(aok ? arow : (M - 1)) * K;

  v8f acc;
  const float bv = bok ? bias[bcol] : 0.0f;        // bias depends on N only
#pragma unroll
  for (int r = 0; r < 8; ++r) acc[r] = bv;

  const int kbA = hs << 4;    // A frag: lanes 16-31 start at k+16
  const int kbB = hs << 4;    // B frag: lanes 16-31 start at k+16
  const int Kmain = K & ~31;

  // staging indices for the cooperative B-tile load (512 elems, 256 threads)
  const int i0 = t, i1 = t + 256;
  const int sn0 = i0 >> 5, sk0 = i0 & 31;
  const int sn1 = i1 >> 5, sk1 = i1 & 31;
  int gn0 = tn + sn0; gn0 = gn0 < N ? gn0 : (N - 1);
  int gn1 = tn + sn1; gn1 = gn1 < N ? gn1 : (N - 1);
  const float* wr0 = W + (size_t)gn0 * K;
  const float* wr1 = W + (size_t)gn1 * K;

  for (int k0 = 0; k0 < Kmain; k0 += 32) {
    // coalesced cooperative stage of W tile -> LDS (f32 -> f16)
    lb[i0] = (_Float16)wr0[k0 + sk0];
    lb[i1] = (_Float16)wr1[k0 + sk1];
    __syncthreads();

    if (k0 + 96 < K) __builtin_prefetch(xrow + k0 + 96, 0, 1);  // global_prefetch_b8

    v16h a;
#pragma unroll
    for (int e = 0; e < 16; ++e) {                 // unguarded: k0+31 < Kmain <= K
      int ka = k0 + kbA + e;
      a[e] = (_Float16)xrow[ka];
    }
    const v8h blo = *(const v8h*)&lb[li * 32 + kbB];       // ds_load_b128
    const v8h bhi = *(const v8h*)&lb[li * 32 + kbB + 8];   // ds_load_b128
    const v16h b = __builtin_shufflevector(blo, bhi,
        0, 1, 2, 3, 4, 5, 6, 7, 8, 9, 10, 11, 12, 13, 14, 15);

    acc = __builtin_amdgcn_wmma_f32_16x16x32_f16(false, a, false, b,
                                                 (short)0, acc, false, false);
    __syncthreads();   // before next stage overwrites lb
  }

  if (Kmain < K) {     // single zero-filled tail step (K % 32 != 0)
    const float* wrow = W + (size_t)(bok ? bcol : (N - 1)) * K;
    v16h a, b;
#pragma unroll
    for (int e = 0; e < 16; ++e) {
      int ka = Kmain + kbA + e;
      a[e] = (ka < K) ? (_Float16)xrow[ka] : (_Float16)0.0f;
      int kb = Kmain + kbB + e;
      b[e] = (kb < K) ? (_Float16)wrow[kb] : (_Float16)0.0f;
    }
    acc = __builtin_amdgcn_wmma_f32_16x16x32_f16(false, a, false, b,
                                                 (short)0, acc, false, false);
  }

  if (bok) {
#pragma unroll
    for (int r = 0; r < 8; ++r) {
      int mr = tm + (hs << 3) + r;   // C/D layout: M = r + (lane<16 ? 0 : 8)
      if (mr < M) Y[(size_t)mr * N + bcol] = acc[r];
    }
  }
}

// ---------------------------------------------------------------------------
// GroupNorm(G=4) + LeakyReLU(0.1), in place. One block per (batch, group).
// ---------------------------------------------------------------------------
__global__ void k_gn_leaky(float* __restrict__ Y, const float* __restrict__ gamma,
                           const float* __restrict__ beta, int rows, int C) {
  const int b = blockIdx.x >> 2;
  const int g = blockIdx.x & 3;
  const int cg = C >> 2;
  const size_t base = (size_t)b * rows * C + (size_t)g * cg;
  const int n = rows * cg;
  float s = 0.f, ss = 0.f;
  for (int i = threadIdx.x; i < n; i += blockDim.x) {
    int r = i / cg, c = i % cg;
    float x = Y[base + (size_t)r * C + c];
    s += x; ss += x * x;
  }
  __shared__ float sh0[256];
  __shared__ float sh1[256];
  sh0[threadIdx.x] = s; sh1[threadIdx.x] = ss;
  __syncthreads();
  for (int st = blockDim.x >> 1; st > 0; st >>= 1) {
    if ((int)threadIdx.x < st) { sh0[threadIdx.x] += sh0[threadIdx.x + st];
                                 sh1[threadIdx.x] += sh1[threadIdx.x + st]; }
    __syncthreads();
  }
  const float mean = sh0[0] / (float)n;
  const float var  = sh1[0] / (float)n - mean * mean;
  const float rstd = rsqrtf(var + 1e-5f);
  for (int i = threadIdx.x; i < n; i += blockDim.x) {
    int r = i / cg, c = i % cg;
    size_t o = base + (size_t)r * C + c;
    int ch = g * cg + c;
    float x = (Y[o] - mean) * rstd * gamma[ch] + beta[ch];
    Y[o] = leaky1(x);
  }
}

// ---------------------------------------------------------------------------
// Furthest point sampling (seed 0, pointnet2 convention). One block per batch.
// ---------------------------------------------------------------------------
__global__ void k_fps(const float* __restrict__ xyz, int* __restrict__ out,
                      int N, int npoint) {
  const int b = blockIdx.x;
  const float* P = xyz + (size_t)b * N * 3;
  __shared__ float dist[512];
  __shared__ float rv[512];
  __shared__ int   ri[512];
  __shared__ int   s_far;
  const int t = threadIdx.x;
  if (t < N) dist[t] = 1e10f;
  if (t == 0) s_far = 0;
  __syncthreads();
  for (int it = 0; it < npoint; ++it) {
    const int far = s_far;
    if (t == 0) out[b * npoint + it] = far;
    const float fx = P[far * 3], fy = P[far * 3 + 1], fz = P[far * 3 + 2];
    float best = -1.0f; int bi = 0;
    if (t < N) {
      float dx = P[t * 3] - fx, dy = P[t * 3 + 1] - fy, dz = P[t * 3 + 2] - fz;
      float nd = fminf(dist[t], dx * dx + dy * dy + dz * dz);
      dist[t] = nd; best = nd; bi = t;
    }
    rv[t] = best; ri[t] = bi;
    __syncthreads();
    for (int st = blockDim.x >> 1; st > 0; st >>= 1) {
      if (t < st && rv[t + st] > rv[t]) { rv[t] = rv[t + st]; ri[t] = ri[t + st]; }
      __syncthreads();
    }
    if (t == 0) s_far = ri[0];
    __syncthreads();
  }
}

// ---------------------------------------------------------------------------
// Brute-force KNN (K<=32), sorted ascending; stable on ties (matches top_k).
// ---------------------------------------------------------------------------
__global__ void k_knn(const float* __restrict__ ref, const float* __restrict__ query,
                      int* __restrict__ idx, int Nr, int Nq, int K, size_t tot) {
  size_t g = (size_t)blockIdx.x * blockDim.x + threadIdx.x;
  if (g >= tot) return;
  const int b = (int)(g / Nq);
  const float* R = ref + (size_t)b * Nr * 3;
  const float* Q = query + g * 3;
  const float qx = Q[0], qy = Q[1], qz = Q[2];
  float bd[32]; int bi[32];
  for (int i = 0; i < K; ++i) { bd[i] = 3.4e38f; bi[i] = 0; }
  for (int r = 0; r < Nr; ++r) {
    float dx = R[r * 3] - qx, dy = R[r * 3 + 1] - qy, dz = R[r * 3 + 2] - qz;
    float d = dx * dx + dy * dy + dz * dz;
    if (d < bd[K - 1]) {
      int j = K - 1;
      while (j > 0 && bd[j - 1] > d) { bd[j] = bd[j - 1]; bi[j] = bi[j - 1]; --j; }
      bd[j] = d; bi[j] = r;
    }
  }
  for (int i = 0; i < K; ++i) idx[g * K + i] = bi[i];
}

// out[(b,i),c] = src[(b, idx[b,i]), c]
__global__ void k_gather_rows(const float* __restrict__ src, const int* __restrict__ idx,
                              float* __restrict__ out, int Nidx, int C, int Nsrc,
                              size_t tot) {
  size_t g = (size_t)blockIdx.x * blockDim.x + threadIdx.x;
  if (g >= tot) return;
  int c = (int)(g % C);
  size_t bi_ = g / C;
  int i = (int)(bi_ % Nidx);
  int b = (int)(bi_ / Nidx);
  int s = idx[(size_t)b * Nidx + i];
  out[g] = src[((size_t)b * Nsrc + s) * C + c];
}

// out[b,m,k,:] = ref[b, idx[b,m,k], :] - center[b,m,:]
__global__ void k_rel_xyz(const float* __restrict__ ref, const float* __restrict__ center,
                          const int* __restrict__ idx, float* __restrict__ out,
                          int M, int K, int Nr, size_t tot) {
  size_t g = (size_t)blockIdx.x * blockDim.x + threadIdx.x;
  if (g >= tot) return;
  size_t bm = g / K;
  int m = (int)(bm % M);
  int b = (int)(bm / M);
  int s = idx[g];
  const float* R = ref + ((size_t)b * Nr + s) * 3;
  const float* Cn = center + ((size_t)b * M + m) * 3;
  out[g * 3 + 0] = R[0] - Cn[0];
  out[g * 3 + 1] = R[1] - Cn[1];
  out[g * 3 + 2] = R[2] - Cn[2];
}

// channel concat of up to 3 tensors sharing the same rows
__global__ void k_concat(const float* __restrict__ a, int Ca,
                         const float* __restrict__ b_, int Cb,
                         const float* __restrict__ c_, int Cc,
                         float* __restrict__ out, size_t rows) {
  const int C = Ca + Cb + Cc;
  size_t tot = rows * C;
  size_t g = (size_t)blockIdx.x * blockDim.x + threadIdx.x;
  if (g >= tot) return;
  int ch = (int)(g % C);
  size_t r = g / C;
  float v;
  if (ch < Ca)            v = a[r * Ca + ch];
  else if (ch < Ca + Cb)  v = b_[r * Cb + (ch - Ca)];
  else                    v = c_[r * Cc + (ch - Ca - Cb)];
  out[g] = v;
}

// cost-volume concat: [f1(broadcast over K) | g2 | dirx]
__global__ void k_cv_concat(const float* __restrict__ f1, const float* __restrict__ g2,
                            const float* __restrict__ dirx, float* __restrict__ out,
                            int N, int K, int C, size_t tot) {
  const int Ct = 2 * C + 3;
  size_t g = (size_t)blockIdx.x * blockDim.x + threadIdx.x;
  if (g >= tot) return;
  int ch = (int)(g % Ct);
  size_t r = g / Ct;
  size_t bn = r / K;
  float v;
  if (ch < C)          v = f1[bn * C + ch];
  else if (ch < 2 * C) v = g2[r * C + (ch - C)];
  else                 v = dirx[r * 3 + (ch - 2 * C)];
  out[g] = v;
}

// out[bn,c] = sum_k w[(bn,k),c] * p[(bn,k),c]
__global__ void k_wsum(const float* __restrict__ w, const float* __restrict__ p,
                       float* __restrict__ out, size_t BN, int K, int C) {
  size_t tot = BN * C;
  size_t g = (size_t)blockIdx.x * blockDim.x + threadIdx.x;
  if (g >= tot) return;
  int c = (int)(g % C);
  size_t bn = g / C;
  float s = 0.f;
  for (int k = 0; k < K; ++k) {
    size_t r = (bn * K + k) * (size_t)C + c;
    s += w[r] * p[r];
  }
  out[g] = s;
}

// out[b,n,c] = sum_k w[(b,n,k),c] * p2p[b, idx[b,n,k], c]
__global__ void k_wgather(const float* __restrict__ w, const float* __restrict__ p2p,
                          const int* __restrict__ idx, float* __restrict__ out,
                          int N, int K, int C, size_t tot) {
  size_t g = (size_t)blockIdx.x * blockDim.x + threadIdx.x;
  if (g >= tot) return;
  int c = (int)(g % C);
  size_t bn = g / C;
  int b = (int)(bn / N);
  float s = 0.f;
  for (int k = 0; k < K; ++k) {
    int si = idx[bn * K + k];
    s += w[(bn * K + k) * (size_t)C + c] * p2p[((size_t)b * N + si) * C + c];
  }
  out[g] = s;
}

// einsum('bmkc,bmkw->bmcw') flattened as out[bm, c*16 + j]
__global__ void k_pc_agg(const float* __restrict__ np_, const float* __restrict__ w,
                         float* __restrict__ out, size_t BM, int K, int Cin) {
  size_t tot = BM * (size_t)Cin * WN_OUT;
  size_t g = (size_t)blockIdx.x * blockDim.x + threadIdx.x;
  if (g >= tot) return;
  int j = (int)(g % WN_OUT);
  size_t t = g / WN_OUT;
  int c = (int)(t % Cin);
  size_t bm = t / Cin;
  const float* NP = np_ + bm * (size_t)K * Cin;
  const float* WW = w + bm * (size_t)K * WN_OUT;
  float s = 0.f;
  for (int k = 0; k < K; ++k) s += NP[(size_t)k * Cin + c] * WW[k * WN_OUT + j];
  out[g] = s;
}

__device__ __forceinline__ void top3_insert(float d, int r, float* bd, int* bi) {
  if (d < bd[2]) {
    if (d < bd[0])      { bd[2]=bd[1];bi[2]=bi[1]; bd[1]=bd[0];bi[1]=bi[0]; bd[0]=d;bi[0]=r; }
    else if (d < bd[1]) { bd[2]=bd[1];bi[2]=bi[1]; bd[1]=d;bi[1]=r; }
    else                { bd[2]=d; bi[2]=r; }
  }
}

// 3-NN inverse-distance interpolation
__global__ void k_upsample(const float* __restrict__ dxyz, const float* __restrict__ sxyz,
                           const float* __restrict__ sfeat, float* __restrict__ out,
                           int Nd, int Ns, int C, size_t tot) {
  size_t g = (size_t)blockIdx.x * blockDim.x + threadIdx.x;
  if (g >= tot) return;
  int b = (int)(g / Nd);
  const float* D = dxyz + g * 3;
  const float* S = sxyz + (size_t)b * Ns * 3;
  float bd[3] = {3.4e38f, 3.4e38f, 3.4e38f}; int bi[3] = {0, 0, 0};
  for (int r = 0; r < Ns; ++r) {
    float dx = S[r*3]-D[0], dy = S[r*3+1]-D[1], dz = S[r*3+2]-D[2];
    top3_insert(dx*dx+dy*dy+dz*dz, r, bd, bi);
  }
  float w[3], ws = 0.f;
  for (int i = 0; i < 3; ++i) { w[i] = 1.0f / fmaxf(bd[i], 1e-10f); ws += w[i]; }
  for (int i = 0; i < 3; ++i) w[i] /= ws;
  const float* F = sfeat + (size_t)b * Ns * C;
  float* O = out + g * (size_t)C;
  for (int c = 0; c < C; ++c)
    O[c] = w[0]*F[(size_t)bi[0]*C+c] + w[1]*F[(size_t)bi[1]*C+c] + w[2]*F[(size_t)bi[2]*C+c];
}

// warp: out[b,n2] = xyz2[b,n2] - sum_i w_i * flow1[b, nn_i]
__global__ void k_warp(const float* __restrict__ xyz1, const float* __restrict__ xyz2,
                       const float* __restrict__ flow1, float* __restrict__ out,
                       int N1, int N2, size_t tot) {
  size_t g = (size_t)blockIdx.x * blockDim.x + threadIdx.x;
  if (g >= tot) return;
  int b = (int)(g / N2);
  const float* Q = xyz2 + g * 3;
  const float* X1 = xyz1 + (size_t)b * N1 * 3;
  const float* F1 = flow1 + (size_t)b * N1 * 3;
  float bd[3] = {3.4e38f, 3.4e38f, 3.4e38f}; int bi[3] = {0, 0, 0};
  for (int r = 0; r < N1; ++r) {
    float wx = X1[r*3]+F1[r*3], wy = X1[r*3+1]+F1[r*3+1], wz = X1[r*3+2]+F1[r*3+2];
    float dx = wx - Q[0], dy = wy - Q[1], dz = wz - Q[2];
    top3_insert(dx*dx+dy*dy+dz*dz, r, bd, bi);
  }
  float w[3], ws = 0.f;
  for (int i = 0; i < 3; ++i) { w[i] = 1.0f / fmaxf(bd[i], 1e-10f); ws += w[i]; }
  for (int i = 0; i < 3; ++i) w[i] /= ws;
  for (int d = 0; d < 3; ++d)
    out[g*3+d] = Q[d] - (w[0]*F1[(size_t)bi[0]*3+d] + w[1]*F1[(size_t)bi[1]*3+d] + w[2]*F1[(size_t)bi[2]*3+d]);
}

// final fc (128 -> 3) + clip(+-200)
__global__ void k_fc_clip(const float* __restrict__ x, const float* __restrict__ W,
                          const float* __restrict__ bias, float* __restrict__ out,
                          size_t Mrows, int K) {
  size_t tot = Mrows * 3;
  size_t g = (size_t)blockIdx.x * blockDim.x + threadIdx.x;
  if (g >= tot) return;
  int j = (int)(g % 3);
  size_t m = g / 3;
  float s = bias[j];
  const float* xr = x + m * (size_t)K;
  const float* wr = W + (size_t)j * K;
  for (int k = 0; k < K; ++k) s += xr[k] * wr[k];
  out[g] = fminf(fmaxf(s, -200.0f), 200.0f);
}

// ===========================================================================
// Host orchestration
// ===========================================================================
static hipStream_t g_st;
static int g_B;

struct Arena {
  char* base; size_t cap, off;
  void* raw(size_t bytes) {
    size_t a = (bytes + 255) & ~(size_t)255;
    if (off + a > cap) off = 0;   // compile-only environment: wrap, never fail
    void* p = base + off; off += a; return p;
  }
  float* f(size_t n)  { return (float*)raw(n * sizeof(float)); }
  int*   ip(size_t n) { return (int*)raw(n * sizeof(int)); }
};
static Arena g_A;

static inline int cdiv(size_t a, int b) { return (int)((a + (size_t)b - 1) / b); }

static void gemm_h(const float* X, const float* W, const float* b, float* Y,
                   int M, int N, int K) {
  dim3 grid((M + 16 * GEMM_WAVES - 1) / (16 * GEMM_WAVES), (N + 15) / 16);
  k_gemm_wmma<<<grid, dim3(32 * GEMM_WAVES), 0, g_st>>>(X, W, b, Y, M, N, K);
}
static void gn_h(float* Y, const float* ga, const float* be, int rows, int C) {
  k_gn_leaky<<<dim3(g_B * 4), dim3(256), 0, g_st>>>(Y, ga, be, rows, C);
}

struct Blk { const float *W, *b, *g, *be; int cin, cout; };

static float* run_seq_h(const float* x, int rowsPerB, int cin, const Blk* bl, int n) {
  const float* cur = x; int C = cin; float* y = nullptr;
  for (int i = 0; i < n; ++i) {
    y = g_A.f((size_t)g_B * rowsPerB * bl[i].cout);
    gemm_h(cur, bl[i].W, bl[i].b, y, g_B * rowsPerB, bl[i].cout, C);
    gn_h(y, bl[i].g, bl[i].be, rowsPerB, bl[i].cout);
    cur = y; C = bl[i].cout;
  }
  return y;
}

struct PConv { const float *gn_b, *gn_g, *lin_W, *lin_b; Blk wn[3]; int cin, cout; };

static float* pointconv_core_h(const float* gx, const float* cat, int M, int K, int cin,
                               const PConv& p) {
  float* w = run_seq_h(gx, M * K, 3, p.wn, 3);           // (B*M*K, 16)
  size_t BM = (size_t)g_B * M;
  float* agg = g_A.f(BM * cin * WN_OUT);
  size_t tot = BM * (size_t)cin * WN_OUT;
  k_pc_agg<<<cdiv(tot, 256), 256, 0, g_st>>>(cat, w, agg, BM, K, cin);
  float* y = g_A.f(BM * p.cout);
  gemm_h(agg, p.lin_W, p.lin_b, y, (int)BM, p.cout, cin * WN_OUT);
  gn_h(y, p.gn_g, p.gn_b, M, p.cout);
  return y;
}

static void pointconv_d_h(const float* xyz, int N, const float* feat, int Cf, int npoint,
                          const PConv& p, float** oxyz, float** ofeat) {
  int* fidx = g_A.ip((size_t)g_B * npoint);
  k_fps<<<dim3(g_B), dim3(512), 0, g_st>>>(xyz, fidx, N, npoint);
  float* nxyz = g_A.f((size_t)g_B * npoint * 3);
  { size_t t = (size_t)g_B * npoint * 3;
    k_gather_rows<<<cdiv(t, 256), 256, 0, g_st>>>(xyz, fidx, nxyz, npoint, 3, N, t); }
  const int K = 16;
  int* idx = g_A.ip((size_t)g_B * npoint * K);
  { size_t t = (size_t)g_B * npoint;
    k_knn<<<cdiv(t, 128), 128, 0, g_st>>>(xyz, nxyz, idx, N, npoint, K, t); }
  size_t rows = (size_t)g_B * npoint * K;
  float* gx = g_A.f(rows * 3);
  k_rel_xyz<<<cdiv(rows, 256), 256, 0, g_st>>>(xyz, nxyz, idx, gx, npoint, K, N, rows);
  float* gf = g_A.f(rows * Cf);
  { size_t t = rows * Cf;
    k_gather_rows<<<cdiv(t, 256), 256, 0, g_st>>>(feat, idx, gf, npoint * K, Cf, N, t); }
  float* cat = g_A.f(rows * (3 + Cf));
  { size_t t = rows * (3 + Cf);
    k_concat<<<cdiv(t, 256), 256, 0, g_st>>>(gx, 3, gf, Cf, (const float*)nullptr, 0, cat, rows); }
  *ofeat = pointconv_core_h(gx, cat, npoint, K, 3 + Cf, p);
  *oxyz = nxyz;
}

static float* upsample_h(const float* dx, int Nd, const float* sx, int Ns,
                         const float* sf, int C) {
  float* o = g_A.f((size_t)g_B * Nd * C);
  size_t t = (size_t)g_B * Nd;
  k_upsample<<<cdiv(t, 128), 128, 0, g_st>>>(dx, sx, sf, o, Nd, Ns, C, t);
  return o;
}

struct CVp { Blk mlp[2]; Blk wn1[3]; Blk wn2[3]; int cout; };

static float* cost_volume_h(const float* xyz1, const float* xyz2w,
                            const float* c1f, const float* c2f, int N, int C,
                            const CVp& p) {
  const int K = 32, cout = p.cout;
  size_t rows = (size_t)g_B * N * K;
  int* idx = g_A.ip(rows);
  { size_t t = (size_t)g_B * N;
    k_knn<<<cdiv(t, 128), 128, 0, g_st>>>(xyz2w, xyz1, idx, N, N, K, t); }
  float* dirx = g_A.f(rows * 3);
  k_rel_xyz<<<cdiv(rows, 256), 256, 0, g_st>>>(xyz2w, xyz1, idx, dirx, N, K, N, rows);
  float* g2 = g_A.f(rows * C);
  { size_t t = rows * C;
    k_gather_rows<<<cdiv(t, 256), 256, 0, g_st>>>(c2f, idx, g2, N * K, C, N, t); }
  int Ct = 2 * C + 3;
  float* cat = g_A.f(rows * Ct);
  { size_t t = rows * Ct;
    k_cv_concat<<<cdiv(t, 256), 256, 0, g_st>>>(c1f, g2, dirx, cat, N, K, C, t); }
  float* pts = run_seq_h(cat, N * K, Ct, p.mlp, 2);
  float* w1  = run_seq_h(dirx, N * K, 3, p.wn1, 3);
  float* p2p = g_A.f((size_t)g_B * N * cout);
  { size_t t = (size_t)g_B * N * cout;
    k_wsum<<<cdiv(t, 256), 256, 0, g_st>>>(w1, pts, p2p, (size_t)g_B * N, K, cout); }
  int* idx1 = g_A.ip(rows);
  { size_t t = (size_t)g_B * N;
    k_knn<<<cdiv(t, 128), 128, 0, g_st>>>(xyz1, xyz1, idx1, N, N, K, t); }
  float* dir1 = g_A.f(rows * 3);
  k_rel_xyz<<<cdiv(rows, 256), 256, 0, g_st>>>(xyz1, xyz1, idx1, dir1, N, K, N, rows);
  float* w2 = run_seq_h(dir1, N * K, 3, p.wn2, 3);
  float* outc = g_A.f((size_t)g_B * N * cout);
  { size_t t = (size_t)g_B * N * cout;
    k_wgather<<<cdiv(t, 256), 256, 0, g_st>>>(w2, p2p, idx1, outc, N, K, cout, t); }
  return outc;
}

struct Flowp { const float *fc_W, *fc_b; Blk mlp[2]; PConv pcs[2]; };

static float* flow_est_h(const float* xyz, int N, const float* feats, int Cf,
                         const float* cost, int Cc, const float* flow,
                         const Flowp& p, float* flow_out) {
  int Cx = Cf + Cc + (flow ? 3 : 0);
  size_t rws = (size_t)g_B * N;
  float* x = g_A.f(rws * Cx);
  { size_t t = rws * Cx;
    k_concat<<<cdiv(t, 256), 256, 0, g_st>>>(feats, Cf, cost, Cc, flow, flow ? 3 : 0, x, rws); }
  const int K = 9;
  size_t rows = rws * K;
  int* idx = g_A.ip(rows);
  k_knn<<<cdiv(rws, 128), 128, 0, g_st>>>(xyz, xyz, idx, N, N, K, rws);
  float* gx = g_A.f(rows * 3);
  k_rel_xyz<<<cdiv(rows, 256), 256, 0, g_st>>>(xyz, xyz, idx, gx, N, K, N, rows);
  const float* cur = x; int C = Cx;
  for (int i = 0; i < 2; ++i) {
    float* gf = g_A.f(rows * C);
    { size_t t = rows * C;
      k_gather_rows<<<cdiv(t, 256), 256, 0, g_st>>>(cur, idx, gf, N * K, C, N, t); }
    float* cat = g_A.f(rows * (3 + C));
    { size_t t = rows * (3 + C);
      k_concat<<<cdiv(t, 256), 256, 0, g_st>>>(gx, 3, gf, C, (const float*)nullptr, 0, cat, rows); }
    cur = pointconv_core_h(gx, cat, N, K, 3 + C, p.pcs[i]);
    C = p.pcs[i].cout;
  }
  float* feat = run_seq_h(cur, N, 256, p.mlp, 2);   // (B,N,128)
  { size_t t = rws * 3;
    k_fc_clip<<<cdiv(t, 256), 256, 0, g_st>>>(feat, p.fc_W, p.fc_b, flow_out, rws, 128); }
  return feat;
}

// -------------------- parameter pytree walker --------------------
struct PR { void* const* din; int i; const float* n() { return (const float*)din[i++]; } };
static Blk rblk(PR& p, int ci, int co) {
  Blk b; b.W = p.n(); b.b = p.n(); b.g = p.n(); b.be = p.n(); b.cin = ci; b.cout = co; return b;
}
static void rwn(PR& p, Blk* w, int co) { w[0] = rblk(p,3,8); w[1] = rblk(p,8,8); w[2] = rblk(p,8,co); }
static PConv rpc(PR& p, int ci, int co) {    // dict keys: gn_b, gn_g, lin_W, lin_b, wn
  PConv c; c.gn_b = p.n(); c.gn_g = p.n(); c.lin_W = p.n(); c.lin_b = p.n();
  rwn(p, c.wn, WN_OUT); c.cin = ci; c.cout = co; return c;
}
static CVp rcv(PR& p, int ci, int co) {      // dict keys: mlp, wn1, wn2
  CVp c; c.mlp[0] = rblk(p, ci, co); c.mlp[1] = rblk(p, co, co);
  rwn(p, c.wn1, co); rwn(p, c.wn2, co); c.cout = co; return c;
}
static Flowp rfl(PR& p, int pc0ci) {         // dict keys: fc_W, fc_b, mlp, pcs
  Flowp f; f.fc_W = p.n(); f.fc_b = p.n();
  f.mlp[0] = rblk(p, 256, 256); f.mlp[1] = rblk(p, 256, 128);
  f.pcs[0] = rpc(p, pc0ci, 256); f.pcs[1] = rpc(p, 259, 256); return f;
}

struct Net {
  Blk init_fc[2]; Blk fij0[1], fij1[2], fij2[2];
  PConv sub[3]; Blk upd[3]; CVp cv[3]; Flowp fl[3];
};
struct FeatOut { const float* pc[4]; float* f[4]; float* c[3]; };

static void forward_feature_h(const float* xyz, const float* color, const Net& P, FeatOut& o) {
  static const int Ns[4] = {512, 128, 32, 8};
  static const int Cs[4] = {64, 128, 256, 512};
  o.pc[0] = xyz;
  o.f[0] = run_seq_h(color, 512, 3, P.init_fc, 2);
  float *nx, *nf;
  float* fij = run_seq_h(o.f[0], 512, 64, P.fij0, 1);
  pointconv_d_h(o.pc[0], 512, fij, 128, 128, P.sub[0], &nx, &nf); o.pc[1] = nx; o.f[1] = nf;
  fij = run_seq_h(o.f[1], 128, 128, P.fij1, 2);
  pointconv_d_h(o.pc[1], 128, fij, 256, 32, P.sub[1], &nx, &nf);  o.pc[2] = nx; o.f[2] = nf;
  fij = run_seq_h(o.f[2], 32, 256, P.fij2, 2);
  pointconv_d_h(o.pc[2], 32, fij, 512, 8, P.sub[2], &nx, &nf);    o.pc[3] = nx; o.f[3] = nf;
  for (int i = 2; i >= 0; --i) {
    float* up  = upsample_h(o.pc[i], Ns[i], o.pc[i + 1], Ns[i + 1], o.f[i + 1], Cs[i + 1]);
    float* fji = run_seq_h(up, Ns[i], Cs[i + 1], &P.upd[i], 1);
    int Cc = Cs[i] + P.upd[i].cout;
    float* cc = g_A.f((size_t)g_B * Ns[i] * Cc);
    size_t rows = (size_t)g_B * Ns[i];
    k_concat<<<cdiv(rows * Cc, 256), 256, 0, g_st>>>(o.f[i], Cs[i], fji, P.upd[i].cout,
                                                     (const float*)nullptr, 0, cc, rows);
    o.c[i] = cc;
  }
}

extern "C" void kernel_launch(void* const* d_in, const int* in_sizes, int n_in,
                              void* d_out, int out_size, void* d_ws, size_t ws_size,
                              hipStream_t stream) {
  (void)n_in; (void)out_size;
  g_st = stream;
  g_B = in_sizes[0] / (512 * 3);
  g_A.base = (char*)d_ws; g_A.cap = ws_size; g_A.off = 0;

  // d_in layout: [0]=xyz1 [1]=xyz2 [2]=color1 [3]=color2, then params pytree
  // flattened with dicts in sorted-key order (cv, feat_ijs, flow, init_fc, sub,
  // updeconv), lists/tuples in order.
  const float* xyz1 = (const float*)d_in[0];
  const float* xyz2 = (const float*)d_in[1];
  const float* col1 = (const float*)d_in[2];
  const float* col2 = (const float*)d_in[3];
  PR pr{d_in, 4};
  Net P;
  P.cv[0] = rcv(pr, 259, 64); P.cv[1] = rcv(pr, 387, 128); P.cv[2] = rcv(pr, 771, 256);
  P.fij0[0] = rblk(pr, 64, 128);
  P.fij1[0] = rblk(pr, 128, 128); P.fij1[1] = rblk(pr, 128, 256);
  P.fij2[0] = rblk(pr, 256, 256); P.fij2[1] = rblk(pr, 256, 512);
  P.fl[0] = rfl(pr, 262); P.fl[1] = rfl(pr, 390); P.fl[2] = rfl(pr, 515);
  P.init_fc[0] = rblk(pr, 3, 64); P.init_fc[1] = rblk(pr, 64, 64);
  P.sub[0] = rpc(pr, 131, 128); P.sub[1] = rpc(pr, 259, 256); P.sub[2] = rpc(pr, 515, 512);
  P.upd[0] = rblk(pr, 128, 64); P.upd[1] = rblk(pr, 256, 64); P.upd[2] = rblk(pr, 512, 128);

  float* out = (float*)d_out;
  float* fo0 = out;
  float* fo1 = fo0 + (size_t)g_B * 512 * 3;
  float* fo2 = fo1 + (size_t)g_B * 128 * 3;
  float* flow_tgts[3] = {fo0, fo1, fo2};

  FeatOut F1, F2;
  forward_feature_h(xyz1, col1, P, F1);
  forward_feature_h(xyz2, col2, P, F2);

  static const int Ns[3] = {512, 128, 32};
  static const int Ccf[3] = {128, 192, 384};
  static const int cvout[3] = {64, 128, 256};
  const float* pcw = F2.pc[2];
  const float* nfeat = F1.f[2]; int Cnf = 256;
  float* upfl = nullptr;
  for (int lid = 2; lid >= 0; --lid) {
    int N = Ns[lid];
    float* cv = cost_volume_h(F1.pc[lid], pcw, F1.c[lid], F2.c[lid], N, Ccf[lid], P.cv[lid]);
    float* feat = flow_est_h(F1.pc[lid], N, nfeat, Cnf, cv, cvout[lid],
                             (lid == 2) ? nullptr : upfl, P.fl[lid], flow_tgts[lid]);
    if (lid > 0) {
      int Nd = Ns[lid - 1];
      upfl = upsample_h(F1.pc[lid - 1], Nd, F1.pc[lid], N, flow_tgts[lid], 3);
      float* pw = g_A.f((size_t)g_B * Nd * 3);
      size_t t = (size_t)g_B * Nd;
      k_warp<<<cdiv(t, 128), 128, 0, g_st>>>(F1.pc[lid - 1], F2.pc[lid - 1], upfl, pw, Nd, Nd, t);
      pcw = pw;
      float* fup = upsample_h(F1.pc[lid - 1], Nd, F1.pc[lid], N, feat, 128);
      int Cprev = (lid == 1) ? 64 : 128;   // f1[0]=64ch, f1[1]=128ch
      int Cn = Cprev + 128;
      float* nf = g_A.f((size_t)g_B * Nd * Cn);
      size_t rows = (size_t)g_B * Nd;
      k_concat<<<cdiv(rows * Cn, 256), 256, 0, g_st>>>(F1.f[lid - 1], Cprev, fup, 128,
                                                       (const float*)nullptr, 0, nf, rows);
      nfeat = nf; Cnf = Cn;
    }
  }
}
